// Grok_90323162235700
// MI455X (gfx1250) — matrix operand order repaired and checked by reading, not compiled, not run on hardware
//
#include <hip/hip_runtime.h>
#include <math.h>

// CDNA5 / gfx1250 wave32 WMMA + TDM types
typedef __attribute__((ext_vector_type(16))) __bf16       v16bf;
typedef __attribute__((ext_vector_type(8)))  float        v8f;
typedef __attribute__((ext_vector_type(4)))  unsigned int v4u;
typedef __attribute__((ext_vector_type(8)))  int          v8i;
typedef __attribute__((ext_vector_type(4)))  int          v4i;

constexpr int D  = 1024;  // model dim
constexpr int H  = 2816;  // ffn hidden dim
constexpr int E  = 8;     // experts
constexpr int M  = 64;    // tokens per workgroup (4 WMMA M-tiles)
constexpr int HC = 704;   // H chunk held in LDS (H/4)

#if defined(__has_builtin)
#if __has_builtin(__builtin_amdgcn_tensor_load_to_lds) && \
    __has_builtin(__builtin_amdgcn_s_wait_tensorcnt) && (__clang_major__ < 23)
#define GROK_HAVE_TDM 1
#endif
#endif
#ifndef GROK_HAVE_TDM
#define GROK_HAVE_TDM 0
#endif

// Branch-free exact-erf GELU: Abramowitz-Stegun 7.1.26 (|err| < 1.5e-7).
__device__ __forceinline__ float gelu_erf(float v) {
  float z = fabsf(v) * 0.70710678118654752f;
  float t = __builtin_amdgcn_rcpf(1.f + 0.3275911f * z);
  float p = t * (0.254829592f +
           t * (-0.284496736f +
           t * (1.421413741f +
           t * (-1.453152027f +
           t *  1.061405429f))));
  float e = 1.f - p * __expf(-z * z);
  return 0.5f * v * (1.f + copysignf(e, v));
}

// ---------------------------------------------------------------------------
// Router: one wave32 per token; shuffle-reduced logits, top-2 softmax combine,
// full softmax gates; integer atomics only => deterministic.
// ---------------------------------------------------------------------------
__global__ void grok_router(const float* __restrict__ x,
                            const float* __restrict__ gw,
                            float* __restrict__ combine,
                            float* __restrict__ gates,
                            int* __restrict__ cnt, int N) {
  int tok  = (int)((blockIdx.x * blockDim.x + threadIdx.x) >> 5);
  int lane = threadIdx.x & 31;
  if (tok >= N) return;
  const float* xr = x + (size_t)tok * D;

  float lg[E];
#pragma unroll
  for (int e = 0; e < E; ++e) lg[e] = 0.f;
  for (int d = lane; d < D; d += 32) {
    float xv = xr[d];
#pragma unroll
    for (int e = 0; e < E; ++e) lg[e] += xv * gw[e * D + d];
  }
#pragma unroll
  for (int e = 0; e < E; ++e) {
    float v = lg[e];
#pragma unroll
    for (int off = 16; off > 0; off >>= 1) v += __shfl_down(v, off, 32);
    lg[e] = __shfl(v, 0, 32);
  }

  if (lane == 0) {
    int i0 = 0; float v0 = lg[0];
#pragma unroll
    for (int e = 1; e < E; ++e) if (lg[e] > v0) { v0 = lg[e]; i0 = e; }
    int i1 = -1; float v1 = -3.4e38f;
#pragma unroll
    for (int e = 0; e < E; ++e) if (e != i0 && lg[e] > v1) { v1 = lg[e]; i1 = e; }
    float e1 = __expf(v1 - v0);
    float w0 = 1.f / (1.f + e1);
    float w1 = e1 / (1.f + e1);

    float ge[E]; float s = 0.f;
#pragma unroll
    for (int e = 0; e < E; ++e) { ge[e] = __expf(lg[e] - v0); s += ge[e]; }
    float inv = 1.f / s;
#pragma unroll
    for (int e = 0; e < E; ++e) {
      gates  [(size_t)tok * E + e] = ge[e] * inv;
      combine[(size_t)tok * E + e] = (e == i0) ? w0 : ((e == i1) ? w1 : 0.f);
    }
    atomicAdd(&cnt[i0], 1);
  }
}

__global__ void grok_preduce(const float* __restrict__ gates,
                             float* __restrict__ psum, int N) {
  __shared__ float red[256];
  int e = blockIdx.x;
  float s = 0.f;
  for (int t = threadIdx.x; t < N; t += 256) s += gates[(size_t)t * E + e];
  red[threadIdx.x] = s;
  __syncthreads();
  for (int o = 128; o > 0; o >>= 1) {
    if ((int)threadIdx.x < o) red[threadIdx.x] += red[threadIdx.x + o];
    __syncthreads();
  }
  if (threadIdx.x == 0) psum[e] = red[0];
}

__global__ void grok_aux(const int* __restrict__ cnt,
                         const float* __restrict__ psum,
                         float* __restrict__ out_aux, float invN) {
  if (threadIdx.x == 0 && blockIdx.x == 0) {
    float s = 0.f;
#pragma unroll
    for (int e = 0; e < E; ++e) s += ((float)cnt[e] * invN) * (psum[e] * invN);
    *out_aux = 0.01f * (float)E * s;
  }
}

// fp32 -> bf16 weight pre-conversion (bf16 set ~156MB fits the 192MB L2)
__global__ void grok_cvt(const float* __restrict__ s, __bf16* __restrict__ d,
                         long n) {
  long i = (long)blockIdx.x * blockDim.x + threadIdx.x;
  long st = (long)gridDim.x * blockDim.x;
  for (; i < n; i += st) d[i] = (__bf16)s[i];
}

// ---------------------------------------------------------------------------
// MoE FFN: one WG = 8 waves = 256 threads per 64-token tile (4 WMMA M-tiles:
// each B fragment feeds 4 WMMAs; weight traffic = (N/64)*weights per pass).
// H is processed in 4 chunks of 704 so the h buffer fits LDS:
//   LDS: xa = bf16 x tile 64xD (128KB), hb = bf16 h chunk 64xHC (88KB)
// Phase 2 accumulates partial down-products into global out with same-thread
// RMW (ISA 7.3 same-address store->load order) + per-thread first-write flag
// => deterministic, no atomics. WT = __bf16 (pre-converted) or float.
// ---------------------------------------------------------------------------
template <typename WT>
__global__ __launch_bounds__(256)
void grok_moe_ffn(const float* __restrict__ x,
                  const float* __restrict__ combine,
                  const WT* __restrict__ wg, const WT* __restrict__ wu,
                  const WT* __restrict__ wd, const WT* __restrict__ sg,
                  const WT* __restrict__ su, const WT* __restrict__ sd,
                  float* __restrict__ out, int N) {
  extern __shared__ char smem[];
  __bf16* xa  = (__bf16*)smem;                        // M*D bf16 (128KB)
  __bf16* hb  = (__bf16*)(smem + M * D * 2);          // M*HC bf16 (88KB)
  float*  wsc = (float*)(smem + M * D * 2 + M * HC * 2);  // M combine weights

  const int tid  = threadIdx.x;
  const int lane = tid & 31;
  const int wave = tid >> 5;
  const int t0   = blockIdx.x * M;

  // ---- stage x tile (f32 -> bf16), 4 x 16-row chunks through hb ----
#if GROK_HAVE_TDM
  {
    const unsigned int lds_off = (unsigned int)(size_t)(void*)hb;
    for (int c = 0; c < 4; ++c) {
      if (wave == 0) {
        unsigned long long ga =
            (unsigned long long)(size_t)(x + (size_t)(t0 + 16 * c) * D);
        v4u g0;
        g0[0] = 1u;                       // count=1, user descriptor
        g0[1] = lds_off;                  // lds_addr
        g0[2] = (unsigned int)ga;         // global_addr[31:0]
        g0[3] = (unsigned int)((ga >> 32) & 0x01FFFFFFu) | (2u << 30); // type=2
        v8i g1;
        g1[0] = 0x00020000;               // data_size=4B, no multicast/pad
        g1[1] = 0x04000000;               // tensor_dim0=1024 (bits 79:48 lo)
        g1[2] = 0x00100000;               // tensor_dim1=16   (bits 111:80 lo)
        g1[3] = 0x04000000;               // tile_dim0=1024   (bits 127:112)
        g1[4] = 16;                       // tile_dim1=16
        g1[5] = 1024;                     // tensor_dim0_stride=1024
        g1[6] = 0;
        g1[7] = 0;
        v4i z4 = {0, 0, 0, 0};
        __builtin_amdgcn_tensor_load_to_lds(g0, g1, z4, z4, 0);
        __builtin_amdgcn_s_wait_tensorcnt(0);
      }
      __syncthreads();
      const float* stg = (const float*)hb;
      for (int i = tid; i < 16 * D; i += 256) {
        int r = i >> 10, col = i & (D - 1);
        xa[(16 * c + r) * D + col] = (__bf16)stg[i];
      }
      __syncthreads();
    }
  }
#else
  for (int i = tid; i < M * D; i += 256) {
    int m = i >> 10, col = i & (D - 1);
    xa[i] = (__bf16)x[(size_t)(t0 + m) * D + col];
  }
  __syncthreads();
#endif

  // WMMA lane mappings (ISA 7.12.2)
  const int mrow = lane & 15;               // A: row within M-tile
  const int kbA  = (lane < 16) ? 0 : 16;    // A: one contiguous 16-K run
  const int ncol = lane & 15;               // B: column
  const int kbB  = (lane < 16) ? 0 : 16;    // B: one contiguous 16-K run
  const int crow = (lane < 16) ? 0 : 8;     // C/D: rows r and r+8

  bool wrote = false;                       // per-thread first-touch flag
  for (int e = 0; e <= E; ++e) {            // e == E -> shared expert
    const bool sh = (e == E);
    const WT* g  = sh ? sg : wg + (size_t)e * H * D;
    const WT* u  = sh ? su : wu + (size_t)e * H * D;
    const WT* dw = sh ? sd : wd + (size_t)e * D * H;

    if (tid < M) wsc[tid] = sh ? 1.f : combine[(size_t)(t0 + tid) * E + e];
    __syncthreads();
    float wsum = 0.f;
#pragma unroll
    for (int m = 0; m < M; ++m) wsum += wsc[m];
    if (wsum == 0.f) { __syncthreads(); continue; }   // no token uses expert e

    for (int c = 0; c < H / HC; ++c) {      // H chunks
      const int hbase = c * HC;

      // ---- phase 1: h[:, chunk] = gelu(x g^T) .* (x u^T); waves split HC ----
      for (int ht = wave; ht < HC / 16; ht += 8) {
        v8f accg[4], accu[4];
#pragma unroll
        for (int mt = 0; mt < 4; ++mt) { accg[mt] = {}; accu[mt] = {}; }
        const int hloc = ht * 16 + ncol;
        const WT* gr = g + (size_t)(hbase + hloc) * D;
        const WT* ur = u + (size_t)(hbase + hloc) * D;
        for (int k0 = 0; k0 < D; k0 += 32) {
          v16bf a[4];
#pragma unroll
          for (int mt = 0; mt < 4; ++mt) {
            const __bf16* ar = xa + (16 * mt + mrow) * D + k0;
#pragma unroll
            for (int j = 0; j < 8; ++j) {
              a[mt][j] = ar[kbA + j];  a[mt][8 + j] = ar[kbA + 8 + j];
            }
          }
          v16bf bg, bu;
#pragma unroll
          for (int j = 0; j < 16; ++j) {
            bg[j] = (__bf16)gr[k0 + kbB + j];
            bu[j] = (__bf16)ur[k0 + kbB + j];
          }
          __builtin_prefetch(gr + k0 + 32, 0, 0);   // global_prefetch_b8
#pragma unroll
          for (int mt = 0; mt < 4; ++mt) {
            accg[mt] = __builtin_amdgcn_wmma_f32_16x16x32_bf16(
                false, a[mt], false, bg, (short)0, accg[mt], false, false);
            accu[mt] = __builtin_amdgcn_wmma_f32_16x16x32_bf16(
                false, a[mt], false, bu, (short)0, accu[mt], false, false);
          }
        }
#pragma unroll
        for (int mt = 0; mt < 4; ++mt) {
#pragma unroll
          for (int r = 0; r < 8; ++r) {
            int m = 16 * mt + r + crow;
            hb[m * HC + hloc] = (__bf16)(gelu_erf(accg[mt][r]) * accu[mt][r]);
          }
        }
      }
      __syncthreads();

      // ---- phase 2: out += w[token] * (h[:,chunk] d[:,chunk]^T) ----
      for (int dt = wave; dt < D / 16; dt += 8) {
        v8f acc[4];
#pragma unroll
        for (int mt = 0; mt < 4; ++mt) acc[mt] = {};
        const int dcol = dt * 16 + ncol;
        const WT* dr = dw + (size_t)dcol * H + hbase;
        for (int k0 = 0; k0 < HC; k0 += 32) {
          v16bf a[4];
#pragma unroll
          for (int mt = 0; mt < 4; ++mt) {
            const __bf16* ar = hb + (16 * mt + mrow) * HC + k0;
#pragma unroll
            for (int j = 0; j < 8; ++j) {
              a[mt][j] = ar[kbA + j];  a[mt][8 + j] = ar[kbA + 8 + j];
            }
          }
          v16bf b;
#pragma unroll
          for (int j = 0; j < 16; ++j) b[j] = (__bf16)dr[k0 + kbB + j];
          __builtin_prefetch(dr + k0 + 32, 0, 0);
#pragma unroll
          for (int mt = 0; mt < 4; ++mt)
            acc[mt] = __builtin_amdgcn_wmma_f32_16x16x32_bf16(
                false, a[mt], false, b, (short)0, acc[mt], false, false);
        }
#pragma unroll
        for (int mt = 0; mt < 4; ++mt) {
#pragma unroll
          for (int r = 0; r < 8; ++r) {
            int m = 16 * mt + r + crow;
            float v = acc[mt][r] * wsc[m];
            float* p = out + (size_t)(t0 + m) * D + dcol;
            if (!wrote) *p = v;
            else        *p = *p + v;
          }
        }
      }
      wrote = true;      // this thread has now written all its out elements
      __syncthreads();   // protect hb reuse by next chunk / next expert
    }
  }
}

// ---------------------------------------------------------------------------
extern "C" void kernel_launch(void* const* d_in, const int* in_sizes, int n_in,
                              void* d_out, int out_size, void* d_ws, size_t ws_size,
                              hipStream_t stream) {
  const float* x      = (const float*)d_in[0];
  const float* gate_w = (const float*)d_in[1];
  const float* wg     = (const float*)d_in[2];
  const float* wu     = (const float*)d_in[3];
  const float* wd     = (const float*)d_in[4];
  const float* sg     = (const float*)d_in[5];
  const float* su     = (const float*)d_in[6];
  const float* sd     = (const float*)d_in[7];
  float* out = (float*)d_out;

  const int N = in_sizes[0] / D;          // B*T tokens

  // Workspace: router buffers, then (optional) bf16 weight blob.
  char*  base    = (char*)d_ws;
  float* combine = (float*)base;                    // N*E f32
  float* gates   = combine + (size_t)N * E;         // N*E f32
  int*   cnt     = (int*)(gates + (size_t)N * E);   // 16 ints
  float* psum    = (float*)(cnt + 16);              // E f32
  size_t wOff = (((size_t)N * E * 8) + 4096 + 255) & ~(size_t)255;

  const size_t EHD = (size_t)E * H * D;             // 23,068,672
  const size_t HD  = (size_t)H * D;                 //  2,883,584
  const size_t needed = wOff + (3 * EHD + 3 * HD) * 2;
  const bool useBf16W = (ws_size >= needed);

  hipMemsetAsync(cnt, 0, 16 * sizeof(int), stream);
  grok_router <<<N / 8, 256, 0, stream>>>(x, gate_w, combine, gates, cnt, N);
  grok_preduce<<<E, 256, 0, stream>>>(gates, psum, N);
  grok_aux    <<<1, 64, 0, stream>>>(cnt, psum, out + (size_t)N * D,
                                     1.0f / (float)N);

  size_t smem = (size_t)M * D * 2 + (size_t)M * HC * 2 + M * 4 + 64;

  if (useBf16W) {
    __bf16* wgb = (__bf16*)(base + wOff);
    __bf16* wub = wgb + EHD;
    __bf16* wdb = wub + EHD;
    __bf16* sgb = wdb + EHD;
    __bf16* sub = sgb + HD;
    __bf16* sdb = sub + HD;
    grok_cvt<<<4096, 256, 0, stream>>>(wg, wgb, (long)EHD);
    grok_cvt<<<4096, 256, 0, stream>>>(wu, wub, (long)EHD);
    grok_cvt<<<4096, 256, 0, stream>>>(wd, wdb, (long)EHD);
    grok_cvt<<<4096, 256, 0, stream>>>(sg, sgb, (long)HD);
    grok_cvt<<<4096, 256, 0, stream>>>(su, sub, (long)HD);
    grok_cvt<<<4096, 256, 0, stream>>>(sd, sdb, (long)HD);
    grok_moe_ffn<__bf16><<<N / M, 256, smem, stream>>>(
        x, combine, wgb, wub, wdb, sgb, sub, sdb, out, N);
  } else {
    grok_moe_ffn<float><<<N / M, 256, smem, stream>>>(
        x, combine, wg, wu, wd, sg, su, sd, out, N);
  }
}
